// TinyModel_22634477650229
// MI455X (gfx1250) — matrix operand order repaired and checked by reading, not compiled, not run on hardware
//
#include <hip/hip_runtime.h>
#include <hip/hip_bf16.h>
#include <math.h>

// ---------------------------------------------------------------------------
// Model dims (fixed by the reference)
// ---------------------------------------------------------------------------
#define NTOK 2048      // B*T
#define DMODEL 1024
#define INDIM 1024
#define FF 4096
#define NLAYER 4
#define NEXP 4
#define NHEAD 16
#define HD 64

typedef __attribute__((ext_vector_type(16))) __bf16 bf16x16;
typedef __attribute__((ext_vector_type(8)))  float  f32x8;

__device__ __forceinline__ __bf16 f2bf(float x) {
  union { float f; unsigned u; } v; v.f = x;
  unsigned r = v.u + 0x7FFFu + ((v.u >> 16) & 1u);   // round-to-nearest-even
  union { unsigned short s; __bf16 b; } o;
  o.s = (unsigned short)(r >> 16);
  return o.b;
}

__device__ __forceinline__ f32x8 f32x8_zero() {
  f32x8 z;
#pragma unroll
  for (int i = 0; i < 8; ++i) z[i] = 0.0f;
  return z;
}

__device__ __forceinline__ f32x8 wmma_bf16(bf16x16 a, bf16x16 b, f32x8 c) {
  return __builtin_amdgcn_wmma_f32_16x16x32_bf16(false, a, false, b, (short)0, c,
                                                 false, false);
}

// A-matrix fragment (16x32 bf16): lane l holds row (l&15);
// elems j -> K = (l>>4)*16 + j  (contiguous 16)
__device__ __forceinline__ bf16x16 load_frag_a(const __bf16* base, int row0,
                                               int stride, int lane) {
  const int r = lane & 15, hs = lane >> 4;
  const __bf16* p = base + (size_t)(row0 + r) * stride + hs * 16;
  bf16x16 f;
#pragma unroll
  for (int j = 0; j < 8; ++j) f[j] = p[j];
#pragma unroll
  for (int j = 0; j < 8; ++j) f[8 + j] = p[8 + j];
  return f;
}

// B-matrix fragment (32x16 bf16): lane l holds column (l&15);
// elems j -> K = (l>>4)*16 + j  (contiguous 16)
__device__ __forceinline__ bf16x16 load_frag_b(const __bf16* base, int row0,
                                               int stride, int lane) {
  const int r = lane & 15, hs = lane >> 4;
  const __bf16* p = base + (size_t)(row0 + r) * stride + hs * 16;
  bf16x16 f;
#pragma unroll
  for (int j = 0; j < 16; ++j) f[j] = p[j];
  return f;
}

// ---------------------------------------------------------------------------
// Generic GEMM:  C[M,N] (+)= act( A[M,K] @ W[N,K]^T ) * rowscale[m]
// 256 threads, macro tile 128x128, K-step 32, double-buffered LDS,
// register-staged global loads, 8 WMMA / wave / K-step, 1 barrier / K-step.
// Requires M%128==0, N%128==0, K%32==0 (true for every GEMM in this model).
// ---------------------------------------------------------------------------
#define BM 128
#define BN 128
#define BK 32
#define LDT 40   // BK + 8 pad: 80-byte row stride keeps 16B alignment

__global__ __launch_bounds__(256) void gemm_xwT(
    const float* __restrict__ A, const float* __restrict__ W,
    float* __restrict__ C, int M, int N, int K,
    const float* __restrict__ rowscale, int rs_stride, int addC, int doGelu) {
  __shared__ __bf16 As[2][BM][LDT];   // 2 x 10 KB
  __shared__ __bf16 Bs[2][BN][LDT];   // 2 x 10 KB

  const int tid = threadIdx.x;
  const int lane = tid & 31;
  const int w = tid >> 5;
  const int wm = (w & 3) * 32;    // 4 waves along M
  const int wn = (w >> 2) * 64;   // 2 waves along N
  const int m0 = blockIdx.y * BM;
  const int n0 = blockIdx.x * BN;

  f32x8 acc[2][4];
#pragma unroll
  for (int i = 0; i < 2; ++i)
#pragma unroll
    for (int j = 0; j < 4; ++j) acc[i][j] = f32x8_zero();

  // Both tiles are 128 rows x 32 cols f32 = 1024 float4 slots -> 4 per thread.
  auto stage_load = [&](int k0, float4* ra, float4* rb) {
#pragma unroll
    for (int it = 0; it < 4; ++it) {
      const int idx = it * 256 + tid;
      const int row = idx >> 3;
      const int c4 = (idx & 7) << 2;
      ra[it] = *(const float4*)(A + (size_t)(m0 + row) * K + k0 + c4);
      rb[it] = *(const float4*)(W + (size_t)(n0 + row) * K + k0 + c4);
    }
  };
  auto stage_store = [&](int buf, const float4* ra, const float4* rb) {
#pragma unroll
    for (int it = 0; it < 4; ++it) {
      const int idx = it * 256 + tid;
      const int row = idx >> 3;
      const int c4 = (idx & 7) << 2;
      As[buf][row][c4 + 0] = f2bf(ra[it].x);
      As[buf][row][c4 + 1] = f2bf(ra[it].y);
      As[buf][row][c4 + 2] = f2bf(ra[it].z);
      As[buf][row][c4 + 3] = f2bf(ra[it].w);
      Bs[buf][row][c4 + 0] = f2bf(rb[it].x);
      Bs[buf][row][c4 + 1] = f2bf(rb[it].y);
      Bs[buf][row][c4 + 2] = f2bf(rb[it].z);
      Bs[buf][row][c4 + 3] = f2bf(rb[it].w);
    }
  };
  auto compute = [&](int buf) {
    bf16x16 af[2], bfr[4];
    af[0] = load_frag_a(&As[buf][0][0], wm + 0, LDT, lane);
    af[1] = load_frag_a(&As[buf][0][0], wm + 16, LDT, lane);
#pragma unroll
    for (int j = 0; j < 4; ++j)
      bfr[j] = load_frag_b(&Bs[buf][0][0], wn + j * 16, LDT, lane);
#pragma unroll
    for (int i = 0; i < 2; ++i)
#pragma unroll
      for (int j = 0; j < 4; ++j) acc[i][j] = wmma_bf16(af[i], bfr[j], acc[i][j]);
  };

  float4 ra[4], rb[4];
  stage_load(0, ra, rb);
  stage_store(0, ra, rb);
  __syncthreads();

  int cur = 0;
  for (int k0 = 0; k0 < K; k0 += BK) {
    const int knext = k0 + BK;
    if (knext < K) {
      stage_load(knext, ra, rb);                   // global loads in flight
      if (knext + BK < K) {                        // keep L2/WGP$ warm
        __builtin_prefetch(A + (size_t)(m0 + (tid >> 1)) * K + knext + BK, 0, 0);
        __builtin_prefetch(W + (size_t)(n0 + (tid >> 1)) * K + knext + BK, 0, 0);
      }
      compute(cur);                                // WMMA on current buffer
      stage_store(cur ^ 1, ra, rb);                // fill the other buffer
      __syncthreads();
      cur ^= 1;
    } else {
      compute(cur);                                // last tile, no staging
    }
  }

  const int col = lane & 15, hs = lane >> 4;
#pragma unroll
  for (int mi = 0; mi < 2; ++mi)
#pragma unroll
    for (int ni = 0; ni < 4; ++ni)
#pragma unroll
      for (int i = 0; i < 8; ++i) {
        const int m = m0 + wm + mi * 16 + i + 8 * hs;
        const int n = n0 + wn + ni * 16 + col;
        float v = acc[mi][ni][i];
        if (doGelu) v = 0.5f * v * (1.0f + erff(v * 0.70710678118654752f));
        if (rowscale) v *= rowscale[(size_t)m * rs_stride];
        float* p = C + (size_t)m * N + n;
        *p = addC ? (*p + v) : v;
      }
}

// ---------------------------------------------------------------------------
// LayerNorm over D=1024, one row per block (256 threads)
// ---------------------------------------------------------------------------
__device__ __forceinline__ float block_sum256(float v, float* sh) {
#pragma unroll
  for (int off = 16; off >= 1; off >>= 1) v += __shfl_xor(v, off);
  int w = threadIdx.x >> 5, lane = threadIdx.x & 31;
  if (lane == 0) sh[w] = v;
  __syncthreads();
  float r = 0.0f;
  if (threadIdx.x < 8) r = sh[threadIdx.x];
  if (threadIdx.x < 8) {
#pragma unroll
    for (int off = 4; off >= 1; off >>= 1) r += __shfl_xor(r, off);
    if (threadIdx.x == 0) sh[0] = r;
  }
  __syncthreads();
  float out = sh[0];
  __syncthreads();
  return out;
}

__global__ __launch_bounds__(256) void ln_k(const float* __restrict__ x,
                                            const float* __restrict__ g,
                                            const float* __restrict__ b,
                                            float* __restrict__ y) {
  __shared__ float sh[8];
  const int row = blockIdx.x;
  const float* xr = x + (size_t)row * DMODEL;
  float s = 0.0f;
  for (int i = threadIdx.x; i < DMODEL; i += 256) s += xr[i];
  float mean = block_sum256(s, sh) * (1.0f / DMODEL);
  float vv = 0.0f;
  for (int i = threadIdx.x; i < DMODEL; i += 256) {
    float d = xr[i] - mean;
    vv += d * d;
  }
  float var = block_sum256(vv, sh) * (1.0f / DMODEL);
  float inv = rsqrtf(var + 1e-5f);
  for (int i = threadIdx.x; i < DMODEL; i += 256)
    y[(size_t)row * DMODEL + i] = (xr[i] - mean) * inv * g[i] + b[i];
}

// ---------------------------------------------------------------------------
// Router: softmax over E=4 logits, top-k, renormalize -> gates[N,4]
// one wave (32 threads) per token
// ---------------------------------------------------------------------------
__global__ __launch_bounds__(32) void router_k(const float* __restrict__ xln,
                                               const float* __restrict__ rw,
                                               float* __restrict__ gates,
                                               const int* __restrict__ topk_p) {
  const int n = blockIdx.x;
  const int lane = threadIdx.x;
  float acc[NEXP] = {0.f, 0.f, 0.f, 0.f};
  for (int d = lane; d < DMODEL; d += 32) {
    float xv = xln[(size_t)n * DMODEL + d];
#pragma unroll
    for (int e = 0; e < NEXP; ++e) acc[e] += xv * rw[(size_t)e * DMODEL + d];
  }
#pragma unroll
  for (int e = 0; e < NEXP; ++e)
#pragma unroll
    for (int off = 16; off >= 1; off >>= 1) acc[e] += __shfl_xor(acc[e], off);
  if (lane == 0) {
    float m = acc[0];
#pragma unroll
    for (int e = 1; e < NEXP; ++e) m = fmaxf(m, acc[e]);
    float p[NEXP], sum = 0.0f;
#pragma unroll
    for (int e = 0; e < NEXP; ++e) {
      p[e] = expf(acc[e] - m);
      sum += p[e];
    }
#pragma unroll
    for (int e = 0; e < NEXP; ++e) p[e] /= sum;
    int k = *topk_p;
    if (k < 1) k = 1;
    if (k > NEXP) k = NEXP;
    float g[NEXP] = {0.f, 0.f, 0.f, 0.f};
    int used[NEXP] = {0, 0, 0, 0};
    float gsum = 0.0f;
    for (int t = 0; t < k; ++t) {
      int bi = -1;
      float bv = -1.0f;
      for (int e = 0; e < NEXP; ++e)
        if (!used[e] && p[e] > bv) { bv = p[e]; bi = e; }
      used[bi] = 1;
      g[bi] = bv;
      gsum += bv;
    }
#pragma unroll
    for (int e = 0; e < NEXP; ++e) gates[(size_t)n * NEXP + e] = g[e] / gsum;
  }
}

// ---------------------------------------------------------------------------
// Attention: one block per (b, h, 16-row q-tile). Two-pass softmax,
// WMMA for QK^T and PV, deterministic 2-way partial reduction for O.
// qkv layout: [NTOK, 3*DMODEL], component c at col c*1024 + h*64 + d
// ---------------------------------------------------------------------------
__global__ __launch_bounds__(256) void attn_k(const float* __restrict__ qkv,
                                              float* __restrict__ ctx) {
  __shared__ float S[16][1024];        // 64 KB raw scores / exp values
  __shared__ __bf16 P[16][1024];       // 32 KB probabilities (bf16)
  __shared__ float Opart[2][16][HD];   // 8 KB deterministic partials

  const int qt = blockIdx.x, h = blockIdx.y, b = blockIdx.z;
  const int tid = threadIdx.x, lane = tid & 31, w = tid >> 5;
  const int r = lane & 15, hs = lane >> 4;

  const size_t rowstride = 3 * DMODEL;
  const float* qbase = qkv + ((size_t)(b * 1024 + qt * 16)) * rowstride + h * HD;
  const float* kbase = qkv + ((size_t)b * 1024) * rowstride + DMODEL + h * HD;
  const float* vbase = qkv + ((size_t)b * 1024) * rowstride + 2 * DMODEL + h * HD;

  // Q fragments: 16 rows x 64 contraction -> 2 A-frags (each wave holds a copy)
  bf16x16 qf[2];
  {
    const float* qr = qbase + (size_t)r * rowstride;
#pragma unroll
    for (int s = 0; s < 2; ++s) {
#pragma unroll
      for (int j = 0; j < 8; ++j) qf[s][j] = f2bf(qr[s * 32 + hs * 16 + j]);
#pragma unroll
      for (int j = 0; j < 8; ++j) qf[s][8 + j] = f2bf(qr[s * 32 + hs * 16 + 8 + j]);
    }
  }

  // Phase 1: S = (Q K^T) * hd^-0.5 ; waves stripe over 64 key tiles
  for (int kt = w; kt < 64; kt += 8) {
    bf16x16 kf[2];
    const float* kr = kbase + (size_t)(kt * 16 + r) * rowstride;
#pragma unroll
    for (int s = 0; s < 2; ++s)
#pragma unroll
      for (int j = 0; j < 16; ++j) kf[s][j] = f2bf(kr[s * 32 + hs * 16 + j]);
    f32x8 acc = f32x8_zero();
    acc = wmma_bf16(qf[0], kf[0], acc);
    acc = wmma_bf16(qf[1], kf[1], acc);
    const int col = kt * 16 + r;
#pragma unroll
    for (int i = 0; i < 8; ++i) S[i + 8 * hs][col] = acc[i] * 0.125f;
  }
  __syncthreads();

  // Phase 2: softmax; wave w owns rows 2w, 2w+1 (deterministic shfl trees)
  for (int rr = w * 2; rr < w * 2 + 2; ++rr) {
    float m = -3.4e38f;
    for (int c = lane; c < 1024; c += 32) m = fmaxf(m, S[rr][c]);
#pragma unroll
    for (int off = 16; off >= 1; off >>= 1) m = fmaxf(m, __shfl_xor(m, off));
    float sum = 0.0f;
    for (int c = lane; c < 1024; c += 32) {
      float e = expf(S[rr][c] - m);
      S[rr][c] = e;
      sum += e;
    }
#pragma unroll
    for (int off = 16; off >= 1; off >>= 1) sum += __shfl_xor(sum, off);
    float inv = 1.0f / sum;
    for (int c = lane; c < 1024; c += 32) P[rr][c] = f2bf(S[rr][c] * inv);
  }
  __syncthreads();

  // Phase 3: O = P @ V. wave w: d-tile (w&3), key-half (w>>2); 16 K-steps each
  {
    const int dt = w & 3, kh = w >> 2;
    f32x8 oacc = f32x8_zero();
    for (int step = 0; step < 16; ++step) {
      const int kb = kh * 512 + step * 32;
      bf16x16 pf = load_frag_a(&P[0][kb], 0, 1024, lane);
      bf16x16 vf;
      const float* vcol = vbase + dt * 16 + r;
#pragma unroll
      for (int j = 0; j < 16; ++j)
        vf[j] = f2bf(vcol[(size_t)(kb + hs * 16 + j) * rowstride]);
      oacc = wmma_bf16(pf, vf, oacc);
    }
#pragma unroll
    for (int i = 0; i < 8; ++i) Opart[kh][i + 8 * hs][dt * 16 + r] = oacc[i];
  }
  __syncthreads();

  // Write ctx[b*1024+qt*16+rr][h*64+d] = sum of the 2 partials (deterministic)
  for (int i = tid; i < 16 * HD; i += 256) {
    int rr = i >> 6, d = i & 63;
    ctx[(size_t)(b * 1024 + qt * 16 + rr) * DMODEL + h * HD + d] =
        Opart[0][rr][d] + Opart[1][rr][d];
  }
}

// ---------------------------------------------------------------------------
// Host-side orchestration
// ---------------------------------------------------------------------------
extern "C" void kernel_launch(void* const* d_in, const int* in_sizes, int n_in,
                              void* d_out, int out_size, void* d_ws, size_t ws_size,
                              hipStream_t stream) {
  const float* x        = (const float*)d_in[0];
  const float* w_in     = (const float*)d_in[1];
  const float* w_out    = (const float*)d_in[2];
  const float* ln1_g    = (const float*)d_in[3];
  const float* ln1_b    = (const float*)d_in[4];
  const float* ln2_g    = (const float*)d_in[5];
  const float* ln2_b    = (const float*)d_in[6];
  const float* qkv_w    = (const float*)d_in[7];
  const float* proj_w   = (const float*)d_in[8];
  const float* router_w = (const float*)d_in[9];
  const float* w1       = (const float*)d_in[10];
  const float* w2       = (const float*)d_in[11];
  const int*   topk     = (const int*)d_in[12];
  float* out = (float*)d_out;

  float* ws = (float*)d_ws;
  float* h     = ws;                                   // [2048,1024]
  float* xln   = h + (size_t)NTOK * DMODEL;            // [2048,1024]
  float* qkv   = xln + (size_t)NTOK * DMODEL;          // [2048,3072]
  float* ctx   = qkv + (size_t)NTOK * 3 * DMODEL;      // [2048,1024]
  float* moeh  = ctx + (size_t)NTOK * DMODEL;          // [2048,4096]
  float* gates = moeh + (size_t)NTOK * FF;             // [2048,4]

  const dim3 blk(256);
  const dim3 gD(DMODEL / BN, NTOK / BM);   // N=1024 GEMMs
  const dim3 gQ(3 * DMODEL / BN, NTOK / BM);
  const dim3 gF(FF / BN, NTOK / BM);

  // h = x @ w_in^T
  gemm_xwT<<<gD, blk, 0, stream>>>(x, w_in, h, NTOK, DMODEL, INDIM,
                                   nullptr, 0, 0, 0);

  for (int l = 0; l < NLAYER; ++l) {
    // --- attention block ---
    ln_k<<<NTOK, blk, 0, stream>>>(h, ln1_g + (size_t)l * DMODEL,
                                   ln1_b + (size_t)l * DMODEL, xln);
    gemm_xwT<<<gQ, blk, 0, stream>>>(xln, qkv_w + (size_t)l * 3 * DMODEL * DMODEL,
                                     qkv, NTOK, 3 * DMODEL, DMODEL,
                                     nullptr, 0, 0, 0);
    attn_k<<<dim3(64, NHEAD, 2), blk, 0, stream>>>(qkv, ctx);
    gemm_xwT<<<gD, blk, 0, stream>>>(ctx, proj_w + (size_t)l * DMODEL * DMODEL,
                                     h, NTOK, DMODEL, DMODEL,
                                     nullptr, 0, /*addC=*/1, 0);

    // --- MoE block ---
    ln_k<<<NTOK, blk, 0, stream>>>(h, ln2_g + (size_t)l * DMODEL,
                                   ln2_b + (size_t)l * DMODEL, xln);
    router_k<<<NTOK, dim3(32), 0, stream>>>(
        xln, router_w + (size_t)l * NEXP * DMODEL, gates, topk);
    for (int e = 0; e < NEXP; ++e) {
      const float* w1e = w1 + ((size_t)l * NEXP + e) * (size_t)FF * DMODEL;
      const float* w2e = w2 + ((size_t)l * NEXP + e) * (size_t)DMODEL * FF;
      gemm_xwT<<<gF, blk, 0, stream>>>(xln, w1e, moeh, NTOK, FF, DMODEL,
                                       nullptr, 0, 0, /*gelu=*/1);
      gemm_xwT<<<gD, blk, 0, stream>>>(moeh, w2e, h, NTOK, DMODEL, FF,
                                       gates + e, NEXP, /*addC=*/1, 0);
    }
  }

  // out = h @ w_out^T
  gemm_xwT<<<gD, blk, 0, stream>>>(h, w_out, out, NTOK, INDIM, DMODEL,
                                   nullptr, 0, 0, 0);
}
